// HashLayerFFN_85753317032356
// MI455X (gfx1250) — matrix operand order repaired and checked by reading, not compiled, not run on hardware
//
#include <hip/hip_runtime.h>

// ---------------- problem constants ----------------
#define BB   4
#define SS   512
#define DD   512      // model dim
#define HH   2048     // hidden dim
#define EE   8        // experts
#define NTOK (BB*SS)  // 2048 tokens
#define LN_EPS 1e-5f

typedef __attribute__((ext_vector_type(16))) __bf16 v16bf;
typedef __attribute__((ext_vector_type(8)))  float  v8f;

// fp32 -> bf16 bits, round-to-nearest-even
__device__ __forceinline__ unsigned short bfbits(float f) {
    union { float f; unsigned u; } v; v.f = f;
    unsigned r = v.u + 0x7FFFu + ((v.u >> 16) & 1u);
    return (unsigned short)(r >> 16);
}

// ---------------- prep kernels ----------------
__global__ void k_zero(int* counts) {
    if (threadIdx.x < 2 * EE) counts[threadIdx.x] = 0;   // counts + cursors
}

__global__ void k_bins(const int* __restrict__ orig, const int* __restrict__ hmap,
                       int* __restrict__ bins, int* __restrict__ counts) {
    int t = blockIdx.x * blockDim.x + threadIdx.x;
    if (t >= NTOK) return;
    int e = hmap[orig[t]];
    bins[t] = e;
    atomicAdd(&counts[e], 1);
}

__global__ void k_scan(const int* __restrict__ counts, int* __restrict__ offsets,
                       int* __restrict__ cursors) {
    if (threadIdx.x == 0) {
        int run = 0;
        for (int e = 0; e < EE; ++e) {
            offsets[e] = run; cursors[e] = run; run += counts[e];
        }
        offsets[EE] = run;
    }
}

__global__ void k_scatter(const int* __restrict__ bins, int* __restrict__ cursors,
                          int* __restrict__ tokmap) {
    int t = blockIdx.x * blockDim.x + threadIdx.x;
    if (t >= NTOK) return;
    int pos = atomicAdd(&cursors[bins[t]], 1);
    tokmap[pos] = t;
}

// x fp32 -> bf16 row-major (one-shot)
__global__ __launch_bounds__(256) void k_cvtX(const float* __restrict__ x,
                                              unsigned short* __restrict__ xbf) {
    int t = blockIdx.x * 256 + threadIdx.x;          // NTOK*DD/4 quads
    float4 v = ((const float4*)x)[t];
    uint2 p;
    p.x = (unsigned)bfbits(v.x) | ((unsigned)bfbits(v.y) << 16);
    p.y = (unsigned)bfbits(v.z) | ((unsigned)bfbits(v.w) << 16);
    ((uint2*)xbf)[t] = p;
}

// B 32x16 (KxN) fragment layout: lane = (n&15) + 16*(k>>4); vgpr j = (k&15)>>1; half = k&1
__device__ __forceinline__ int bIdx(int k, int n) {
    int ln = (n & 15) + ((k >> 4) << 4);
    int j  = (k & 15) >> 1;
    return ln * 16 + j * 2 + (k & 1);
}
// A-fragment write helper for an aligned group of 4 K values (c4 % 4 == 0):
// lane = m + 16*(K>>4); vgpr j = (K&15)>>1; dword pair write
__device__ __forceinline__ int aQuadDword(int row, int c4) {
    int ln = row + ((c4 >> 4) << 4);
    int j  = (c4 & 15) >> 1;
    return ln * 8 + j;   // dword index within a 512-half kt tile
}

// Convert W[E][Kdim][Ndim] fp32 -> bf16 in per-lane WMMA B-fragment layout:
// Wf[e][ns][kt][lane(32)][16 halves]; grid (EE*Kdim/32, Ndim/512), block 256.
__global__ __launch_bounds__(256) void k_cvtW(const float* __restrict__ W,
                                              unsigned short* __restrict__ Wf,
                                              int Kdim, int Ndim) {
    __shared__ __align__(64) unsigned short lB[32 * 512];
    const int ktiles = Kdim >> 5;
    const int e  = blockIdx.x / ktiles;
    const int kt = blockIdx.x % ktiles;
    const int k0 = kt << 5;
    const int n0 = blockIdx.y << 9;
    const int tid = threadIdx.x;
    const float* We = W + (size_t)e * Kdim * Ndim;

#pragma unroll
    for (int i = 0; i < 16; ++i) {
        int lin = i * 1024 + tid * 4;
        int k = lin >> 9, nl = lin & 511;
        float4 wv = *(const float4*)(We + (size_t)(k0 + k) * Ndim + n0 + nl);
        float f[4] = { wv.x, wv.y, wv.z, wv.w };
#pragma unroll
        for (int q = 0; q < 4; ++q) {
            int n = nl + q;
            lB[(n >> 4) * 512 + bIdx(k, n)] = bfbits(f[q]);
        }
    }
    __syncthreads();
    int sb = tid >> 3, part = tid & 7;
    size_t outHalf = (((size_t)e * (Ndim >> 4) + (n0 >> 4) + sb) * ktiles + kt) * 512 + part * 64;
    uint4* dst = (uint4*)(Wf + outHalf);
    const uint4* src = (const uint4*)(lB + sb * 512 + part * 64);
#pragma unroll
    for (int i = 0; i < 8; ++i) dst[i] = src[i];
}

// ---------------- GEMM1: h = relu(x_gathered @ W1[e] + b1[e]) -> bf16 ----------------
// grid: (EE*128, HH/512), block 256 (8 waves). Tile: M=16 tokens, N=512.
// Entire A row-block (16 x 512 bf16 = 16KB) staged to LDS once -> barrier-free K loop.
__global__ __launch_bounds__(256) void k_ffn1(
    const unsigned short* __restrict__ xbf, const unsigned short* __restrict__ W1f,
    const float* __restrict__ b1,
    const int* __restrict__ counts, const int* __restrict__ offsets,
    const int* __restrict__ tokmap, unsigned short* __restrict__ hbuf)
{
    __shared__ __align__(64) unsigned short lA[16 * 512];   // [kt(16)][lane(32)][16 halves]
    __shared__ int stok[16];

    const int e   = blockIdx.x >> 7;
    const int mt  = blockIdx.x & 127;
    const int cnt = counts[e];
    const int rs  = mt * 16;
    if (rs >= cnt) return;
    const int off = offsets[e];
    const int g0  = off + rs;
    const int n0  = blockIdx.y << 9;
    const int tid = threadIdx.x;
    const int lane = tid & 31, w = tid >> 5;

    if (tid < 16) {
        int gi = g0 + tid, gmax = off + cnt - 1;
        stok[tid] = tokmap[gi > gmax ? gmax : gi];
    }
    __syncthreads();

    // stage all A fragments: 2048 quads (16 rows x 128 quads)
#pragma unroll
    for (int i = 0; i < 8; ++i) {
        int q = tid + i * 256;
        int row = q >> 7, c4 = (q & 127) * 4;
        uint2 p = *(const uint2*)(xbf + (size_t)stok[row] * DD + c4);
        int kt = c4 >> 5, kk = c4 & 31;
        *(uint2*)((unsigned*)lA + kt * 256 + aQuadDword(row, kk)) = p;
    }
    __syncthreads();

    const v16bf* BF = (const v16bf*)W1f;   // [e][ns][kt][lane]
    const v16bf* Af = (const v16bf*)lA;
    v8f acc[4] = {};

#pragma unroll 4
    for (int kt = 0; kt < (DD >> 5); ++kt) {
        v16bf a = Af[kt * 32 + lane];
#pragma unroll
        for (int s = 0; s < 4; ++s) {
            int ns = (n0 >> 4) + w * 4 + s;
            v16bf b = BF[((size_t)(e * 128 + ns) * (DD >> 5) + kt) * 32 + lane];
            acc[s] = __builtin_amdgcn_wmma_f32_16x16x32_bf16(
                false, a, false, b, (short)0, acc[s], false, false);
        }
    }

    // epilogue: +b1, ReLU, store bf16 h (guard tail rows of this expert)
#pragma unroll
    for (int s = 0; s < 4; ++s) {
        int ncol = n0 + w * 64 + s * 16 + (lane & 15);
        float bb = b1[e * HH + ncol];
#pragma unroll
        for (int v = 0; v < 8; ++v) {
            int m = ((lane >> 4) << 3) + v;
            if (rs + m < cnt) {
                float hv = acc[s][v] + bb;
                hv = hv > 0.f ? hv : 0.f;
                hbuf[(size_t)(g0 + m) * HH + ncol] = bfbits(hv);
            }
        }
    }
}

// ---------------- GEMM2 + residual + LayerNorm + scatter ----------------
// grid: EE*128, block 256. Tile: M=16 x N=512 (full D), K=2048.
// Entire A row-block (16 x 2048 bf16 = 64KB) staged once -> barrier-free K loop.
__global__ __launch_bounds__(256) void k_ffn2(
    const float* __restrict__ x, const unsigned short* __restrict__ W2f,
    const float* __restrict__ b2,
    const float* __restrict__ gamma, const float* __restrict__ beta,
    const int* __restrict__ counts, const int* __restrict__ offsets,
    const int* __restrict__ tokmap, const unsigned short* __restrict__ hbuf,
    float* __restrict__ out)
{
    __shared__ __align__(64) unsigned short lA[64 * 512];   // [kt(64)][lane(32)][16 halves]
    __shared__ int stok[16];
    __shared__ float pp[16][128];
    __shared__ float qq[16][128];
    __shared__ float smu[16], srstd[16];

    const int e   = blockIdx.x >> 7;
    const int mt  = blockIdx.x & 127;
    const int cnt = counts[e];
    const int rs  = mt * 16;
    if (rs >= cnt) return;
    const int off = offsets[e];
    const int g0  = off + rs;
    const int tid = threadIdx.x;
    const int lane = tid & 31, w = tid >> 5;

    if (tid < 16) {
        int gi = g0 + tid, gmax = off + cnt - 1;
        stok[tid] = tokmap[gi > gmax ? gmax : gi];
    }
    __syncthreads();

    // stage all A fragments from bf16 h: 8192 quads (16 rows x 512 quads)
#pragma unroll
    for (int i = 0; i < 32; ++i) {
        int q = tid + i * 256;
        int row = q >> 9, c4 = (q & 511) * 4;
        int gr = g0 + row; if (gr > NTOK - 1) gr = NTOK - 1;   // clamp OOB tail
        uint2 p = *(const uint2*)(hbuf + (size_t)gr * HH + c4);
        int kt = c4 >> 5, kk = c4 & 31;
        *(uint2*)((unsigned*)lA + kt * 256 + aQuadDword(row, kk)) = p;
    }
    __syncthreads();

    const v16bf* BF = (const v16bf*)W2f;   // [e][ns(32)][kt(64)][lane]
    const v16bf* Af = (const v16bf*)lA;
    v8f acc[4] = {};

#pragma unroll 4
    for (int kt = 0; kt < (HH >> 5); ++kt) {
        v16bf a = Af[kt * 32 + lane];
#pragma unroll
        for (int s = 0; s < 4; ++s) {
            int ns = w * 4 + s;
            v16bf b = BF[((size_t)(e * 32 + ns) * (HH >> 5) + kt) * 32 + lane];
            acc[s] = __builtin_amdgcn_wmma_f32_16x16x32_bf16(
                false, a, false, b, (short)0, acc[s], false, false);
        }
    }

    // epilogue: y = acc + b2 + x  (registers), then deterministic LN
    float yv[4][8];
#pragma unroll
    for (int s = 0; s < 4; ++s) {
        int ncol = w * 64 + s * 16 + (lane & 15);
        float bb = b2[e * DD + ncol];
#pragma unroll
        for (int v = 0; v < 8; ++v) {
            int m = ((lane >> 4) << 3) + v;
            yv[s][v] = acc[s][v] + bb + x[(size_t)stok[m] * DD + ncol];
        }
    }
    int cid = w * 16 + (lane & 15);
#pragma unroll
    for (int v = 0; v < 8; ++v) {
        int m = ((lane >> 4) << 3) + v;
        float ps = 0.f, pq = 0.f;
#pragma unroll
        for (int s = 0; s < 4; ++s) { ps += yv[s][v]; pq += yv[s][v] * yv[s][v]; }
        pp[m][cid] = ps; qq[m][cid] = pq;
    }
    __syncthreads();
    if (tid < 16) {
        float s1 = 0.f, s2 = 0.f;
        for (int i = 0; i < 128; ++i) { s1 += pp[tid][i]; s2 += qq[tid][i]; }
        float mu  = s1 * (1.f / DD);
        float var = s2 * (1.f / DD) - mu * mu;
        smu[tid] = mu; srstd[tid] = rsqrtf(var + LN_EPS);
    }
    __syncthreads();
#pragma unroll
    for (int s = 0; s < 4; ++s) {
        int ncol = w * 64 + s * 16 + (lane & 15);
        float g = gamma[ncol], bt = beta[ncol];
#pragma unroll
        for (int v = 0; v < 8; ++v) {
            int m = ((lane >> 4) << 3) + v;
            if (rs + m < cnt) {
                float o = (yv[s][v] - smu[m]) * srstd[m] * g + bt;
                out[(size_t)stok[m] * DD + ncol] = o;
            }
        }
    }
}

// ---------------- host entry ----------------
extern "C" void kernel_launch(void* const* d_in, const int* in_sizes, int n_in,
                              void* d_out, int out_size, void* d_ws, size_t ws_size,
                              hipStream_t stream) {
    const float* x     = (const float*)d_in[0];
    const float* W1    = (const float*)d_in[1];
    const float* b1    = (const float*)d_in[2];
    const float* W2    = (const float*)d_in[3];
    const float* b2    = (const float*)d_in[4];
    const float* gamma = (const float*)d_in[5];
    const float* beta  = (const float*)d_in[6];
    const int*   orig  = (const int*)d_in[7];
    const int*   hmap  = (const int*)d_in[8];
    float* out = (float*)d_out;

    // workspace layout (MB-aligned regions)
    char* ws = (char*)d_ws;
    int* counts  = (int*)ws;              // 8
    int* cursors = counts + 8;            // 8
    int* offsets = cursors + 8;           // 9 (+pad)
    int* bins    = offsets + 16;          // 2048
    int* tokmap  = bins + NTOK;           // 2048
    unsigned short* hbuf = (unsigned short*)(ws + (1ull << 20));            // 8 MB
    unsigned short* xbf  = (unsigned short*)(ws + (9ull << 20));            // 2 MB
    unsigned short* W1f  = (unsigned short*)(ws + (11ull << 20));           // 16 MB
    unsigned short* W2f  = (unsigned short*)(ws + (27ull << 20));           // 16 MB

    k_zero<<<1, 32, 0, stream>>>(counts);
    k_bins<<<NTOK / 256, 256, 0, stream>>>(orig, hmap, bins, counts);
    k_scan<<<1, 32, 0, stream>>>(counts, offsets, cursors);
    k_scatter<<<NTOK / 256, 256, 0, stream>>>(bins, cursors, tokmap);
    k_cvtX<<<(NTOK * DD / 4) / 256, 256, 0, stream>>>(x, xbf);
    k_cvtW<<<dim3(EE * (DD / 32), HH / 512), 256, 0, stream>>>(W1, W1f, DD, HH);
    k_cvtW<<<dim3(EE * (HH / 32), DD / 512), 256, 0, stream>>>(W2, W2f, HH, DD);
    k_ffn1<<<dim3(EE * 128, HH / 512), 256, 0, stream>>>(xbf, W1f, b1, counts, offsets, tokmap, hbuf);
    k_ffn2<<<EE * 128, 256, 0, stream>>>(x, W2f, b2, gamma, beta, counts, offsets, tokmap, hbuf, out);
}
